// MultiHeadedAttention_56152402428003
// MI455X (gfx1250) — matrix operand order repaired and checked by reading, not compiled, not run on hardware
//
#include <hip/hip_runtime.h>
#include <hip/hip_bf16.h>

#define Bn 8
#define Sn 1024
#define Dn 768
#define Hn 12
#define HDn 64

typedef __attribute__((ext_vector_type(16))) _Float16 v16h;
typedef __attribute__((ext_vector_type(8)))  float    v8f;

union Frag {
  v16h v;
  _Float16 h[16];
  int4 q[2];
};

__device__ __forceinline__ void cvt8(const float* __restrict__ p, _Float16* d) {
  float4 a = *(const float4*)p;
  float4 b = *(const float4*)(p + 4);
  d[0] = (_Float16)a.x; d[1] = (_Float16)a.y; d[2] = (_Float16)a.z; d[3] = (_Float16)a.w;
  d[4] = (_Float16)b.x; d[5] = (_Float16)b.y; d[6] = (_Float16)b.z; d[7] = (_Float16)b.w;
}

__device__ __forceinline__ v8f wmma_f16(const Frag& a, const Frag& b, v8f c) {
  return __builtin_amdgcn_wmma_f32_16x16x32_f16(false, a.v, false, b.v, (short)0, c,
                                                false, false);
}

// ---------------------------------------------------------------------------
// Projection: Y = X @ W^T + bias for q/k/v (blockIdx.z selects which).
//   z==0 -> qh[b][h][s][hd] (pre-scaled by 1/sqrt(HD))
//   z==1 -> kh[b][h][s][hd]
//   z==2 -> vT[b][h][hd][s] (transposed so P@V B-fragments are contiguous)
// Block: 128 thr (4 waves), each wave a 16(M)x64(N) tile; block tile 64x64.
// ---------------------------------------------------------------------------
__global__ __launch_bounds__(128) void proj_kernel(
    const float* __restrict__ Xq, const float* __restrict__ Xk, const float* __restrict__ Xv,
    const float* __restrict__ Wqp, const float* __restrict__ Wkp, const float* __restrict__ Wvp,
    const float* __restrict__ bqp, const float* __restrict__ bkp, const float* __restrict__ bvp,
    _Float16* __restrict__ qh, _Float16* __restrict__ kh, _Float16* __restrict__ vT)
{
  const int z = blockIdx.z;
  const float* X    = (z == 0) ? Xq  : (z == 1) ? Xk  : Xv;
  const float* W    = (z == 0) ? Wqp : (z == 1) ? Wkp : Wvp;
  const float* bias = (z == 0) ? bqp : (z == 1) ? bkp : bvp;

  const int wave = threadIdx.x >> 5;
  const int lane = threadIdx.x & 31;
  const int ln   = lane & 15;
  const bool hiL = lane >= 16;

  const int mBase = blockIdx.x * 64 + wave * 16;
  const int nBase = blockIdx.y * 64;

  v8f acc[4] = {};

  const float* xrow = X + (size_t)(mBase + ln) * Dn;
  for (int k0 = 0; k0 < Dn; k0 += 32) {
    Frag a;
    const int kb = k0 + (hiL ? 8 : 0);      // A layout: K chunks {kb..kb+7, kb+16..kb+23}
    cvt8(xrow + kb, a.h);
    cvt8(xrow + kb + 16, a.h + 8);
#pragma unroll
    for (int t = 0; t < 4; ++t) {
      const int n = nBase + t * 16 + ln;    // B lane = output dim n, K contiguous
      const float* wrow = W + (size_t)n * Dn;
      const int kwb = k0 + (hiL ? 8 : 0);
      Frag bf;
      cvt8(wrow + kwb, bf.h);
      cvt8(wrow + kwb + 16, bf.h + 8);
      acc[t] = wmma_f16(a, bf, acc[t]);
    }
  }

  // C/D layout: lane -> col n = lane&15; VGPR r -> row m = r + (lane>=16 ? 8 : 0)
  const int mT = mBase + (hiL ? 8 : 0);
  const int bb = mT >> 10;           // / S
  const int s0 = mT & (Sn - 1);
#pragma unroll
  for (int t = 0; t < 4; ++t) {
    const int n = nBase + t * 16 + ln;
    const float bsv = bias[n];
    const int hh = n >> 6;
    const int hd = n & (HDn - 1);
    if (z == 2) {
      union { _Float16 hh8[8]; int4 qq; } tmp;
#pragma unroll
      for (int r = 0; r < 8; ++r) tmp.hh8[r] = (_Float16)(acc[t][r] + bsv);
      const size_t idx = (((size_t)bb * Hn + hh) * HDn + hd) * Sn + s0;
      *(int4*)(vT + idx) = tmp.qq;   // 8 consecutive s, 16B aligned
    } else {
      _Float16* dst = (z == 0) ? qh : kh;
      const float scl = (z == 0) ? 0.125f : 1.0f;   // fold 1/sqrt(64) into q
#pragma unroll
      for (int r = 0; r < 8; ++r) {
        const size_t idx = (((size_t)bb * Hn + hh) * Sn + (s0 + r)) * HDn + hd;
        dst[idx] = (_Float16)((acc[t][r] + bsv) * scl);
      }
    }
  }
}

// ---------------------------------------------------------------------------
// Fused attention: one block per (b, h, 16-query tile).
// scores(WMMA) -> LDS -> dual softmax + combine -> prob store + LDS f16
// -> out = P @ V (WMMA, waves split K) -> ds_add_f32 reduce -> store.
// ---------------------------------------------------------------------------
__global__ __launch_bounds__(256) void attn_kernel(
    const _Float16* __restrict__ qh, const _Float16* __restrict__ kh,
    const _Float16* __restrict__ vT,
    const float* __restrict__ rel, const unsigned char* __restrict__ mask,
    const float* __restrict__ l1p,
    float* __restrict__ out, float* __restrict__ prob)
{
  extern __shared__ char smem[];
  float    (*sc)[Sn]    = (float(*)[Sn])smem;                              // 64 KB
  _Float16 (*pr)[Sn]    = (_Float16(*)[Sn])(smem + 16 * Sn * 4);           // 32 KB
  float    (*oAcc)[HDn] = (float(*)[HDn])(smem + 16 * Sn * 4 + 16 * Sn * 2); // 4 KB

  const int b  = blockIdx.z, h = blockIdx.y;
  const int q0 = blockIdx.x * 16;
  const size_t bh = (size_t)b * Hn + h;

  const int tid  = threadIdx.x;
  const int wave = tid >> 5;
  const int lane = tid & 31;
  const int ln   = lane & 15;
  const bool hiL = lane >= 16;
  const int akb  = hiL ? 8 : 0;    // A-fragment K sub-offset
  const int bkb  = hiL ? 8 : 0;    // B-fragment K sub-offset

  for (int i = tid; i < 16 * HDn; i += 256) ((float*)oAcc)[i] = 0.f;

  // ---- scores = (q/8) @ k^T : wave w covers key columns [w*128, w*128+128) ----
  const _Float16* qrow = qh + (bh * Sn + q0 + ln) * HDn;
  Frag a0, a1;                      // K = hd, two 32-wide steps
  a0.q[0] = *(const int4*)(qrow + akb);
  a0.q[1] = *(const int4*)(qrow + akb + 16);
  a1.q[0] = *(const int4*)(qrow + 32 + akb);
  a1.q[1] = *(const int4*)(qrow + 32 + akb + 16);

#pragma unroll
  for (int t = 0; t < 8; ++t) {
    const int key = wave * 128 + t * 16 + ln;
    const _Float16* krow = kh + (bh * Sn + key) * HDn;
    Frag b0, b1;
    b0.q[0] = *(const int4*)(krow + bkb);
    b0.q[1] = *(const int4*)(krow + bkb + 16);
    b1.q[0] = *(const int4*)(krow + 32 + bkb);
    b1.q[1] = *(const int4*)(krow + 32 + bkb + 16);
    v8f c = {};
    c = wmma_f16(a0, b0, c);
    c = wmma_f16(a1, b1, c);
    const int col = wave * 128 + t * 16 + ln;
    const int rb  = hiL ? 8 : 0;
#pragma unroll
    for (int r = 0; r < 8; ++r) sc[rb + r][col] = c[r];
  }
  __syncthreads();

  // ---- dual softmax, 16 threads per row, shfl_xor(width=16) reductions ----
  const int row = tid >> 4;
  const int j   = tid & 15;
  const float l1 = *l1p;
  const float* relrow = rel + ((size_t)b * Sn + q0 + row) * Sn;
  const unsigned char* mrow = mask + ((size_t)b * Sn + q0 + row) * Sn;

  float smax = -3.4e38f, rmax = -3.4e38f;
  for (int i = 0; i < Sn / 16; ++i) {
    const int c = j + i * 16;
    const bool mm = mrow[c] != 0;
    float s = sc[row][c];
    if (mm) s = -1e9f;
    sc[row][c] = s;
    smax = fmaxf(smax, s);
    const float rraw = mm ? relrow[c] : 0.f;   // relm = rel * mask
    const float rv = (rraw == 0.f) ? -1e4f : rraw;
    pr[row][c] = (_Float16)rv;
    rmax = fmaxf(rmax, rv);
  }
#pragma unroll
  for (int m = 8; m >= 1; m >>= 1) {
    smax = fmaxf(smax, __shfl_xor(smax, m, 16));
    rmax = fmaxf(rmax, __shfl_xor(rmax, m, 16));
  }
  float ssum = 0.f, rsum = 0.f;
  for (int i = 0; i < Sn / 16; ++i) {
    const int c = j + i * 16;
    const float es = __expf(sc[row][c] - smax);
    ssum += es; sc[row][c] = es;
    const float er = __expf((float)pr[row][c] - rmax);
    rsum += er; pr[row][c] = (_Float16)er;
  }
#pragma unroll
  for (int m = 8; m >= 1; m >>= 1) {
    ssum += __shfl_xor(ssum, m, 16);
    rsum += __shfl_xor(rsum, m, 16);
  }
  const float sinv = (1.f - l1) / ssum;
  const float rinv = l1 / rsum;
  float* probrow = prob + (bh * Sn + q0 + row) * Sn;
  for (int i = 0; i < Sn / 16; ++i) {
    const int c = j + i * 16;
    const float p = sc[row][c] * sinv + (float)pr[row][c] * rinv;
    probrow[c] = p;                 // required 402MB output, coalesced
    pr[row][c] = (_Float16)p;       // f16 copy for the P@V WMMA
  }
  __syncthreads();

  // ---- out = P @ V : wave w covers keys [w*128, w*128+128) ----
  v8f oc[4] = {};
#pragma unroll
  for (int ks = 0; ks < 4; ++ks) {
    const int kbase = wave * 128 + ks * 32;
    Frag a;                                        // A from LDS prob (f16)
    a.q[0] = *(const int4*)&pr[ln][kbase + akb];
    a.q[1] = *(const int4*)&pr[ln][kbase + akb + 16];
#pragma unroll
    for (int nt = 0; nt < 4; ++nt) {
      const int hd = nt * 16 + ln;
      const _Float16* vrow = vT + (bh * HDn + hd) * Sn + kbase;  // contiguous in K
      Frag bf;
      bf.q[0] = *(const int4*)(vrow + bkb);
      bf.q[1] = *(const int4*)(vrow + bkb + 16);
      oc[nt] = wmma_f16(a, bf, oc[nt]);
    }
  }
  {
    const int rb = hiL ? 8 : 0;
#pragma unroll
    for (int nt = 0; nt < 4; ++nt) {
      const int hd = nt * 16 + ln;
#pragma unroll
      for (int r = 0; r < 8; ++r)
        atomicAdd(&oAcc[rb + r][hd], oc[nt][r]);   // ds_add_f32 cross-wave reduce
    }
  }
  __syncthreads();
  for (int i = tid; i < 16 * HDn; i += 256) {
    const int r = i >> 6, hd = i & 63;
    out[((size_t)b * Sn + q0 + r) * Dn + h * HDn + hd] = oAcc[r][hd];
  }
}

extern "C" void kernel_launch(void* const* d_in, const int* in_sizes, int n_in,
                              void* d_out, int out_size, void* d_ws, size_t ws_size,
                              hipStream_t stream) {
  const float* query = (const float*)d_in[0];
  const float* key   = (const float*)d_in[1];
  const float* value = (const float*)d_in[2];
  const float* rel   = (const float*)d_in[3];
  const unsigned char* mask = (const unsigned char*)d_in[4];
  const float* l1    = (const float*)d_in[5];
  const float* Wq    = (const float*)d_in[6];
  const float* bq    = (const float*)d_in[7];
  const float* Wk    = (const float*)d_in[8];
  const float* bk    = (const float*)d_in[9];
  const float* Wv    = (const float*)d_in[10];
  const float* bv    = (const float*)d_in[11];

  float* out  = (float*)d_out;
  float* prob = out + (size_t)Bn * Sn * Dn;     // tuple order: (out, prob_attn)

  const size_t nProj = (size_t)Bn * Hn * Sn * HDn;  // 6.29M f16 each
  _Float16* qh = (_Float16*)d_ws;
  _Float16* kh = qh + nProj;
  _Float16* vT = kh + nProj;

  dim3 pg(Bn * Sn / 64, Dn / 64, 3);
  proj_kernel<<<pg, 128, 0, stream>>>(query, key, value, Wq, Wk, Wv,
                                      bq, bk, bv, qh, kh, vT);

  dim3 ag(Sn / 16, Hn, Bn);
  const size_t shmem = (size_t)16 * Sn * 4 + (size_t)16 * Sn * 2 + (size_t)16 * HDn * 4;
  attn_kernel<<<ag, 256, shmem, stream>>>(qh, kh, vT, rel, mask, l1, out, prob);
}
